// Tmix_2353642078795
// MI455X (gfx1250) — matrix operand order repaired and checked by reading, not compiled, not run on hardware
//
#include <hip/hip_runtime.h>
#include <math.h>

// Problem constants (match reference)
#define B_ 4
#define T_ 1024
#define C_ 1024
#define H_ 16
#define HS_ 64
#define M_ (B_ * T_)  // 4096 rows in all GEMMs

#if defined(__has_builtin)
# if __has_builtin(__builtin_amdgcn_global_load_async_to_lds_b128)
#  define USE_ASYNC_LDS 1
# endif
#endif
#ifndef USE_ASYNC_LDS
# define USE_ASYNC_LDS 0
#endif

typedef __attribute__((ext_vector_type(16))) __bf16 v16bf;
typedef __attribute__((ext_vector_type(8)))  float v8f;
#if USE_ASYNC_LDS
typedef int v4i __attribute__((vector_size(16)));
typedef __attribute__((address_space(1))) void gvoid_t;
typedef __attribute__((address_space(3))) void lvoid_t;
typedef __attribute__((address_space(1))) v4i g_v4i;
typedef __attribute__((address_space(3))) v4i l_v4i;
#endif

static __device__ __forceinline__ unsigned short f2bf(float f) {
  unsigned int u = __float_as_uint(f);
  u += 0x7fffu + ((u >> 16) & 1u);   // round-to-nearest-even
  return (unsigned short)(u >> 16);
}
static __device__ __forceinline__ __bf16 bfbits(unsigned short s) {
  union { unsigned short u; __bf16 b; } c; c.u = s; return c.b;
}

static __device__ __forceinline__ void wait_async0() {
#if USE_ASYNC_LDS
# if __has_builtin(__builtin_amdgcn_s_wait_asynccnt)
  __builtin_amdgcn_s_wait_asynccnt(0);
# else
  asm volatile("s_wait_asynccnt 0x0" ::: "memory");
# endif
#endif
}

// ---------------------------------------------------------------------------
// GEMM: C[M x N](f32) = A[M x K](bf16) @ Wt[N x K](bf16, pre-transposed).
// Block = 256 threads (8 waves), tile 64x64, K-step 32, double-buffered LDS,
// tile copies via GLOBAL_LOAD_ASYNC_TO_LDS_B128 (fallback: plain b128 copy).
// Requires: N % 64 == 0, K % 32 == 0 (true for all call sites here).
// ---------------------------------------------------------------------------
__global__ __launch_bounds__(256)
void gemm_bf16_wmma(const unsigned short* __restrict__ A,
                    const unsigned short* __restrict__ Wt,
                    float* __restrict__ C, int M, int N, int K) {
  __shared__ __align__(16) unsigned short As[2][64][40];
  __shared__ __align__(16) unsigned short Ws[2][64][40];
  const int t    = threadIdx.x;
  const int lane = t & 31;
  const int wv   = t >> 5;
  const int m0   = blockIdx.y * 64;
  const int n0   = blockIdx.x * 64;

  // Tile copy mapping: thread -> (row 0..63, k-chunk of 8 bf16 = 16B)
  const int rl = t >> 2;
  const int kc = (t & 3) * 8;
  const unsigned short* gA = A  + (size_t)(m0 + rl) * K + kc;
  const unsigned short* gW = Wt + (size_t)(n0 + rl) * K + kc;

  auto issue = [&](int buf, int k0) {
#if USE_ASYNC_LDS
    __builtin_amdgcn_global_load_async_to_lds_b128(
        (g_v4i*)(gvoid_t*)(void*)(size_t)(gA + k0),
        (l_v4i*)(lvoid_t*)(void*)&As[buf][rl][kc], 0, 0);
    __builtin_amdgcn_global_load_async_to_lds_b128(
        (g_v4i*)(gvoid_t*)(void*)(size_t)(gW + k0),
        (l_v4i*)(lvoid_t*)(void*)&Ws[buf][rl][kc], 0, 0);
#else
    *(uint4*)&As[buf][rl][kc] = *(const uint4*)(gA + k0);
    *(uint4*)&Ws[buf][rl][kc] = *(const uint4*)(gW + k0);
#endif
  };

  v8f acc0 = {}; v8f acc1 = {};
  const int aRow   = (wv >> 1) * 16 + (lane & 15);   // A-frag row
  const int kHalfA = (lane >= 16) ? 8 : 0;           // A-frag K split (ISA 7.12.2)
  const int nloc   = (wv & 1) * 32 + (lane & 15);    // B-frag column
  const int kHalfB = (lane >= 16) ? 8 : 0;           // B-frag K split

  issue(0, 0);
  wait_async0();
  __syncthreads();
  int buf = 0;
  for (int k0 = 0; k0 < K; k0 += 32) {
    if (k0 + 32 < K) issue(buf ^ 1, k0 + 32);   // overlap copy with WMMA

    v16bf a, b0, b1;
    #pragma unroll
    for (int j = 0; j < 8; ++j) {
      int kp = 2 * j + ((j >= 4) ? 8 : 0) + kHalfA;
      a[2 * j]     = bfbits(As[buf][aRow][kp]);
      a[2 * j + 1] = bfbits(As[buf][aRow][kp + 1]);
    }
    #pragma unroll
    for (int i = 0; i < 16; ++i) {
      b0[i] = bfbits(Ws[buf][nloc][kHalfB + i + ((i >= 8) ? 8 : 0)]);
      b1[i] = bfbits(Ws[buf][nloc + 16][kHalfB + i + ((i >= 8) ? 8 : 0)]);
    }
    acc0 = __builtin_amdgcn_wmma_f32_16x16x32_bf16(false, a, false, b0, (short)0, acc0, false, false);
    acc1 = __builtin_amdgcn_wmma_f32_16x16x32_bf16(false, a, false, b1, (short)0, acc1, false, false);

    wait_async0();
    __syncthreads();
    buf ^= 1;
  }

  // D layout: VGPR r -> M = r (lanes 0-15) / 8+r (lanes 16-31); N = lane&15
  int mBase = m0 + (wv >> 1) * 16 + ((lane >= 16) ? 8 : 0);
  int nA = n0 + (wv & 1) * 32 + (lane & 15);
  #pragma unroll
  for (int rI = 0; rI < 8; ++rI) {
    C[(size_t)(mBase + rI) * N + nA]      = acc0[rI];
    C[(size_t)(mBase + rI) * N + nA + 16] = acc1[rI];
  }
}

// ---------------------------------------------------------------------------
// Weight transpose + fp32->bf16 (one-time per call): Wt[n*K+k] = bf16(W[k*N+n])
// Tiled 32x32 through LDS so both global accesses are coalesced.
// ---------------------------------------------------------------------------
__global__ __launch_bounds__(256)
void wtrans_kernel(const float* __restrict__ W, unsigned short* __restrict__ Wt,
                   int K, int N) {
  __shared__ float tile[32][33];
  int n0 = blockIdx.x * 32, k0 = blockIdx.y * 32;
  int tx = threadIdx.x & 31, ty = threadIdx.x >> 5;  // 8 rows per pass
  #pragma unroll
  for (int i = 0; i < 32; i += 8)
    tile[ty + i][tx] = W[(size_t)(k0 + ty + i) * N + (n0 + tx)];
  __syncthreads();
  #pragma unroll
  for (int i = 0; i < 32; i += 8)
    Wt[(size_t)(n0 + ty + i) * K + (k0 + tx)] = f2bf(tile[tx][ty + i]);
}

// ---------------------------------------------------------------------------
// Elementwise kernels (activation producers emit bf16 for GEMM A operands)
// ---------------------------------------------------------------------------
__global__ __launch_bounds__(256)
void prelude_kernel(const float* __restrict__ x,
                    const float* __restrict__ mr, const float* __restrict__ mw,
                    const float* __restrict__ mk, const float* __restrict__ mv,
                    const float* __restrict__ ma, const float* __restrict__ mg,
                    unsigned short* __restrict__ xr, unsigned short* __restrict__ xw,
                    unsigned short* __restrict__ xk, unsigned short* __restrict__ xv,
                    unsigned short* __restrict__ xa, unsigned short* __restrict__ xg) {
  size_t idx = (size_t)blockIdx.x * blockDim.x + threadIdx.x;
  int c = (int)(idx & (C_ - 1));
  size_t tok = idx >> 10;
  int tpos = (int)(tok & (T_ - 1));
  float xc = x[idx];
  float prev = (tpos > 0) ? x[idx - C_] : 0.f;
  float xx = prev - xc;                   // time_shift(x) - x
  xr[idx] = f2bf(xc + xx * mr[c]);
  xw[idx] = f2bf(xc + xx * mw[c]);
  xk[idx] = f2bf(xc + xx * mk[c]);
  xv[idx] = f2bf(xc + xx * mv[c]);
  xa[idx] = f2bf(xc + xx * ma[c]);
  xg[idx] = f2bf(xc + xx * mg[c]);
}

__global__ __launch_bounds__(256)
void tanh_bf16_kernel(const float* __restrict__ in, unsigned short* __restrict__ out, int n) {
  int i = blockIdx.x * blockDim.x + threadIdx.x;
  if (i < n) out[i] = f2bf(tanhf(in[i]));
}

__global__ __launch_bounds__(256)
void sigmoid_bf16_kernel(const float* __restrict__ in, unsigned short* __restrict__ out, int n) {
  int i = blockIdx.x * blockDim.x + threadIdx.x;
  if (i < n) out[i] = f2bf(1.f / (1.f + expf(-in[i])));
}

__global__ __launch_bounds__(256)
void f2bf_kernel(const float* __restrict__ in, unsigned short* __restrict__ out, int n) {
  int i = blockIdx.x * blockDim.x + threadIdx.x;
  if (i < n) out[i] = f2bf(in[i]);
}

// a = sigmoid(bias[c] + z), in place, fp32 (feeds kprep)
__global__ __launch_bounds__(256)
void sigmoid_bias_kernel(float* __restrict__ z, const float* __restrict__ bias) {
  size_t idx = (size_t)blockIdx.x * blockDim.x + threadIdx.x;
  int c = (int)(idx & (C_ - 1));
  z[idx] = 1.f / (1.f + expf(-(bias[c] + z[idx])));
}

// wd = exp(-exp(-softplus(-(w0 + z)) - 0.5))  (the WKV per-channel decay)
__global__ __launch_bounds__(256)
void wdecay_kernel(float* __restrict__ z, const float* __restrict__ w0) {
  size_t idx = (size_t)blockIdx.x * blockDim.x + threadIdx.x;
  int c = (int)(idx & (C_ - 1));
  float u = w0[c] + z[idx];
  float w = -log1pf(expf(-u)) - 0.5f;
  z[idx] = expf(-expf(w));
}

// Per (token, head): kk = normalize(k * k_k); akk = -kk; bb = kk * a;
// k <- k * (1 + (a-1) * k_a)
__global__ __launch_bounds__(256)
void kprep_kernel(const float* __restrict__ a, const float* __restrict__ kkw,
                  const float* __restrict__ kaw, float* __restrict__ k,
                  float* __restrict__ akk, float* __restrict__ bb) {
  int gid = blockIdx.x * blockDim.x + threadIdx.x;
  int head = gid & (H_ - 1);
  int tok  = gid >> 4;
  size_t base = (size_t)tok * C_ + head * HS_;
  int cb = head * HS_;
  float ss = 0.f;
  for (int i = 0; i < HS_; ++i) {
    float tv = k[base + i] * kkw[cb + i];
    ss += tv * tv;
  }
  float inv = 1.f / fmaxf(sqrtf(ss), 1e-12f);
  for (int i = 0; i < HS_; ++i) {
    float kv = k[base + i];
    float av = a[base + i];
    float kkn = kv * kkw[cb + i] * inv;
    akk[base + i] = -kkn;
    bb[base + i]  = kkn * av;
    k[base + i]   = kv * (1.f + (av - 1.f) * kaw[cb + i]);
  }
}

// GroupNorm(per head, eps) + r*k*r_k bonus + gate; emits bf16 for Wo GEMM
__global__ __launch_bounds__(256)
void post_kernel(const float* __restrict__ y, const float* __restrict__ r,
                 const float* __restrict__ k, const float* __restrict__ v,
                 const float* __restrict__ g, const float* __restrict__ r_k,
                 const float* __restrict__ lnw, const float* __restrict__ lnb,
                 unsigned short* __restrict__ yg) {
  int gid = blockIdx.x * blockDim.x + threadIdx.x;
  int head = gid & (H_ - 1);
  int tok  = gid >> 4;
  size_t base = (size_t)tok * C_ + head * HS_;
  int cb = head * HS_;
  float mu = 0.f, m2 = 0.f, dot = 0.f;
  for (int i = 0; i < HS_; ++i) {
    float yv = y[base + i];
    mu += yv; m2 += yv * yv;
    dot += r[base + i] * k[base + i] * r_k[cb + i];
  }
  mu *= (1.f / HS_);
  float var = m2 * (1.f / HS_) - mu * mu;
  float inv = rsqrtf(var + 0.00064f);
  for (int i = 0; i < HS_; ++i) {
    float yn = (y[base + i] - mu) * inv * lnw[cb + i] + lnb[cb + i]
             + dot * v[base + i];
    yg[base + i] = f2bf(yn * g[base + i]);
  }
}

// ---------------------------------------------------------------------------
// WKV scan: one 64-thread block per (batch, head). Thread j owns row j of the
// 64x64 state in VGPRs; per-step vectors broadcast via double-buffered LDS.
// S = S*diag(w) + (S a) b^T + v k^T ;  y = S r
// ---------------------------------------------------------------------------
__global__ __launch_bounds__(64)
void wkv_kernel(const float* __restrict__ r, const float* __restrict__ wd,
                const float* __restrict__ k, const float* __restrict__ v,
                const float* __restrict__ akk, const float* __restrict__ bb,
                float* __restrict__ y) {
  __shared__ float lR[2][HS_], lW[2][HS_], lK[2][HS_],
                   lV[2][HS_], lA[2][HS_], lB[2][HS_];
  int bh = blockIdx.x;
  int batch = bh >> 4;
  int head  = bh & (H_ - 1);
  int j = threadIdx.x;
  float S[HS_];
  #pragma unroll
  for (int n = 0; n < HS_; ++n) S[n] = 0.f;
  size_t base = (size_t)batch * T_ * C_ + head * HS_ + j;
  for (int t = 0; t < T_; ++t) {
    int pb = t & 1;
    size_t off = base + (size_t)t * C_;
    lR[pb][j] = r[off];  lW[pb][j] = wd[off];  lK[pb][j] = k[off];
    lV[pb][j] = v[off];  lA[pb][j] = akk[off]; lB[pb][j] = bb[off];
    if (t + 32 < T_) {   // hide latency of the next steps' vector loads
      size_t pf = off + (size_t)32 * C_;
      __builtin_prefetch(&r[pf], 0, 3);  __builtin_prefetch(&wd[pf], 0, 3);
      __builtin_prefetch(&k[pf], 0, 3);  __builtin_prefetch(&v[pf], 0, 3);
      __builtin_prefetch(&akk[pf], 0, 3); __builtin_prefetch(&bb[pf], 0, 3);
    }
    __syncthreads();
    float sa = 0.f;
    #pragma unroll
    for (int n = 0; n < HS_; ++n) sa += S[n] * lA[pb][n];
    float vj = lV[pb][j];
    float yj = 0.f;
    #pragma unroll
    for (int n = 0; n < HS_; ++n) {
      float s = S[n] * lW[pb][n] + sa * lB[pb][n] + vj * lK[pb][n];
      S[n] = s;
      yj += s * lR[pb][n];
    }
    y[off] = yj;
  }
}

// ---------------------------------------------------------------------------
// Host-side launch
// ---------------------------------------------------------------------------
extern "C" void kernel_launch(void* const* d_in, const int* in_sizes, int n_in,
                              void* d_out, int out_size, void* d_ws, size_t ws_size,
                              hipStream_t stream) {
  (void)in_sizes; (void)n_in; (void)out_size; (void)ws_size;
  const float* x    = (const float*)d_in[0];
  const float* x_r  = (const float*)d_in[1];
  const float* x_w  = (const float*)d_in[2];
  const float* x_k  = (const float*)d_in[3];
  const float* x_v  = (const float*)d_in[4];
  const float* x_a  = (const float*)d_in[5];
  const float* x_g  = (const float*)d_in[6];
  const float* w0   = (const float*)d_in[7];
  const float* w1   = (const float*)d_in[8];
  const float* w2   = (const float*)d_in[9];
  const float* a0   = (const float*)d_in[10];
  const float* a1   = (const float*)d_in[11];
  const float* a2   = (const float*)d_in[12];
  // d_in[13..15] = v0,v1,v2: unused — first-layer v-mix is exactly identity
  const float* g1w  = (const float*)d_in[16];
  const float* g2w  = (const float*)d_in[17];
  const float* k_k  = (const float*)d_in[18];
  const float* k_a  = (const float*)d_in[19];
  const float* r_k  = (const float*)d_in[20];
  const float* Wr   = (const float*)d_in[21];
  const float* Wk   = (const float*)d_in[22];
  const float* Wv   = (const float*)d_in[23];
  const float* Wo   = (const float*)d_in[24];
  const float* ln_w = (const float*)d_in[25];
  const float* ln_b = (const float*)d_in[26];
  float* out = (float*)d_out;

  // Workspace: 9 big fp32 slots + low-rank fp32 temps + bf16 region (~158 MB)
  float* ws = (float*)d_ws;
  const size_t BIG = (size_t)M_ * C_;
  float* S0 = ws + 0 * BIG;  float* S1 = ws + 1 * BIG;  float* S2 = ws + 2 * BIG;
  float* S3 = ws + 3 * BIG;  float* S4 = ws + 4 * BIG;  float* S5 = ws + 5 * BIG;
  float* S6 = ws + 6 * BIG;  float* S7 = ws + 7 * BIG;  float* S8 = ws + 8 * BIG;
  float* T1f = ws + 9 * BIG;                        // M_ x 64  (fp32)
  float* T2f = T1f + (size_t)M_ * 64;               // M_ x 128 (fp32)
  unsigned short* T1b = (unsigned short*)(T2f + (size_t)M_ * 128);
  unsigned short* T2b = T1b + (size_t)M_ * 64;
  unsigned short* WtR = T2b + (size_t)M_ * 128;
  unsigned short* WtK = WtR + (size_t)C_ * C_;
  unsigned short* WtV = WtK + (size_t)C_ * C_;
  unsigned short* WtO = WtV + (size_t)C_ * C_;
  unsigned short* W1t = WtO + (size_t)C_ * C_;      // 64  x 1024
  unsigned short* W2t = W1t + (size_t)64 * C_;      // 1024 x 64
  unsigned short* A1t = W2t + (size_t)C_ * 64;
  unsigned short* A2t = A1t + (size_t)64 * C_;
  unsigned short* G1t = A2t + (size_t)C_ * 64;      // 128 x 1024
  unsigned short* G2t = G1t + (size_t)128 * C_;     // 1024 x 128
  // bf16 aliases of big slots (activation GEMM A operands)
  unsigned short* X0 = (unsigned short*)S0;  unsigned short* X1 = (unsigned short*)S1;
  unsigned short* X2 = (unsigned short*)S2;  unsigned short* X3 = (unsigned short*)S3;
  unsigned short* X4 = (unsigned short*)S4;  unsigned short* X5 = (unsigned short*)S5;
  unsigned short* YG = (unsigned short*)S0;

  dim3 blk(256);
  dim3 gN1024(C_ / 64, M_ / 64);
  dim3 gN64(1, M_ / 64);
  dim3 gN128(2, M_ / 64);
  const int nBig = (M_ * C_) / 256;
  const int nTH  = (M_ * H_) / 256;

  // 0) one-time weight transpose+convert to bf16 (Wt[N][K])
  wtrans_kernel<<<dim3(C_/32, C_/32), blk, 0, stream>>>(Wr, WtR, C_, C_);
  wtrans_kernel<<<dim3(C_/32, C_/32), blk, 0, stream>>>(Wk, WtK, C_, C_);
  wtrans_kernel<<<dim3(C_/32, C_/32), blk, 0, stream>>>(Wv, WtV, C_, C_);
  wtrans_kernel<<<dim3(C_/32, C_/32), blk, 0, stream>>>(Wo, WtO, C_, C_);
  wtrans_kernel<<<dim3(64/32,  C_/32), blk, 0, stream>>>(w1,  W1t, C_, 64);
  wtrans_kernel<<<dim3(C_/32,  64/32), blk, 0, stream>>>(w2,  W2t, 64, C_);
  wtrans_kernel<<<dim3(64/32,  C_/32), blk, 0, stream>>>(a1,  A1t, C_, 64);
  wtrans_kernel<<<dim3(C_/32,  64/32), blk, 0, stream>>>(a2,  A2t, 64, C_);
  wtrans_kernel<<<dim3(128/32, C_/32), blk, 0, stream>>>(g1w, G1t, C_, 128);
  wtrans_kernel<<<dim3(C_/32, 128/32), blk, 0, stream>>>(g2w, G2t, 128, C_);
  // 1) token-shift mixing -> bf16 xr..xg
  prelude_kernel<<<nBig, blk, 0, stream>>>(x, x_r, x_w, x_k, x_v, x_a, x_g,
                                           X0, X1, X2, X3, X4, X5);
  // 2) big projections
  gemm_bf16_wmma<<<gN1024, blk, 0, stream>>>(X0, WtR, S6, M_, C_, C_);  // r
  gemm_bf16_wmma<<<gN1024, blk, 0, stream>>>(X2, WtK, S7, M_, C_, C_);  // k
  gemm_bf16_wmma<<<gN1024, blk, 0, stream>>>(X3, WtV, S2, M_, C_, C_);  // v
  // 3) decay: wd = exp(-exp(-softplus(-(w0 + tanh(xw@w1)@w2)) - 0.5))
  gemm_bf16_wmma<<<gN64, blk, 0, stream>>>(X1, W1t, T1f, M_, 64, C_);
  tanh_bf16_kernel<<<(M_ * 64) / 256, blk, 0, stream>>>(T1f, T1b, M_ * 64);
  gemm_bf16_wmma<<<gN1024, blk, 0, stream>>>(T1b, W2t, S3, M_, C_, 64);
  wdecay_kernel<<<nBig, blk, 0, stream>>>(S3, w0);
  // 4) a = sigmoid(a0 + xa@a1@a2)
  gemm_bf16_wmma<<<gN64, blk, 0, stream>>>(X4, A1t, T1f, M_, 64, C_);
  f2bf_kernel<<<(M_ * 64) / 256, blk, 0, stream>>>(T1f, T1b, M_ * 64);
  gemm_bf16_wmma<<<gN1024, blk, 0, stream>>>(T1b, A2t, S1, M_, C_, 64);
  sigmoid_bias_kernel<<<nBig, blk, 0, stream>>>(S1, a0);
  // 5) g = sigmoid(xg@g1)@g2
  gemm_bf16_wmma<<<gN128, blk, 0, stream>>>(X5, G1t, T2f, M_, 128, C_);
  sigmoid_bf16_kernel<<<(M_ * 128) / 256, blk, 0, stream>>>(T2f, T2b, M_ * 128);
  gemm_bf16_wmma<<<gN1024, blk, 0, stream>>>(T2b, G2t, S4, M_, C_, 128);
  // 6) kk normalize + k update: akk -> S0, bb -> S5, k updated in S7
  kprep_kernel<<<nTH, blk, 0, stream>>>(S1, k_k, k_a, S7, S0, S5);
  // 7) WKV recurrence: y -> S8
  wkv_kernel<<<B_ * H_, dim3(64), 0, stream>>>(S6, S3, S7, S2, S0, S5, S8);
  // 8) GroupNorm + bonus + gate -> bf16 yg (over dead akk slot)
  post_kernel<<<nTH, blk, 0, stream>>>(S8, S6, S7, S2, S4, r_k, ln_w, ln_b, YG);
  // 9) output projection
  gemm_bf16_wmma<<<gN1024, blk, 0, stream>>>(YG, WtO, out, M_, C_, C_);
}
